// GATv2_84782654423396
// MI455X (gfx1250) — matrix operand order repaired and checked by reading, not compiled, not run on hardware
//
#include <hip/hip_runtime.h>
#include <hip/hip_bf16.h>
#include <math.h>

// ---------------------------------------------------------------------------
// GATv2 2-layer forward for MI455X (gfx1250, wave32, WMMA).
// GEMMs use v_wmma_f32_16x16x32_bf16 (bf16 A/B, f32 accumulate).
// Edge softmax/aggregation use L2-resident gathers + float atomics.
// ---------------------------------------------------------------------------

typedef __bf16 bf16_t;
typedef bf16_t v16bf __attribute__((ext_vector_type(16)));
typedef float  v8f   __attribute__((ext_vector_type(8)));

#define NEG_SLOPE 0.2f

__device__ __forceinline__ float lrelu(float x) { return x > 0.f ? x : NEG_SLOPE * x; }

// float atomic-max via signed/unsigned int compare trick (init memory to -inf)
__device__ __forceinline__ void atomicMaxF(float* addr, float v) {
    if (v >= 0.f) atomicMax((int*)addr, __float_as_int(v));
    else          atomicMin((unsigned int*)addr, __float_as_uint(v));
}

// ---- WMMA operand loaders (documented CDNA5 16-bit layouts) ---------------
// A (16x32, MxK): lane L: M = L&15, half = L>>4.
//   elements 0..7  -> K = k0 + half*8 + i
//   elements 8..15 -> K = k0 + 16 + half*8 + (i-8)
__device__ __forceinline__ v16bf load_A(const float* __restrict__ A, int row, int lda,
                                        int k0, int half) {
    const float* p0 = A + (size_t)row * lda + k0 + half * 8;
    v16bf a;
#pragma unroll
    for (int i = 0; i < 8; ++i) a[i] = (bf16_t)p0[i];
#pragma unroll
    for (int i = 0; i < 8; ++i) a[8 + i] = (bf16_t)p0[16 + i];
    return a;
}

// B (32x16, KxN): lane L: N = L&15, half = L>>4. element i<8 -> K = k0 + half*8 + i, i>=8 -> K = k0 + 16 + half*8 + (i-8)
__device__ __forceinline__ v16bf load_B(const float* __restrict__ B, int ldb,
                                        int k0, int n0, int lane) {
    const int half = lane >> 4;
    const float* p = B + (size_t)(k0 + half * 8) * ldb + n0 + (lane & 15);
    v16bf b;
#pragma unroll
    for (int i = 0; i < 16; ++i) b[i] = (bf16_t)p[(size_t)(i + (i & 8)) * ldb];
    return b;
}

// D (16x16 f32): VGPR r: M = r + 8*half, N = lane&15
__device__ __forceinline__ void store_D(float* __restrict__ D, int ldd,
                                        int m0, int n0, int lane, v8f c) {
    const int half = lane >> 4;
    const int ncol = n0 + (lane & 15);
#pragma unroll
    for (int r = 0; r < 8; ++r)
        D[(size_t)(m0 + r + half * 8) * ldd + ncol] = c[r];
}

__device__ __forceinline__ v8f wmma_bf16(v16bf a, v16bf b, v8f c) {
    return __builtin_amdgcn_wmma_f32_16x16x32_bf16(false, a, false, b, (short)0, c,
                                                   false, false);
}

// ---- workspace init (must run every launch: atomics accumulate) -----------
__global__ __launch_bounds__(256) void init_ws(float* out1, float* den1, float* m1,
                                               float* out2, float* den2, float* m2,
                                               int N) {
    size_t t = (size_t)blockIdx.x * 256 + threadIdx.x;
    size_t n128 = (size_t)N * 128;
    if (t < n128) out1[t] = 0.f;
    if (t < (size_t)N * 16) out2[t] = 0.f;
    if (t < (size_t)N * 2) { den1[t] = 0.f; m1[t] = -INFINITY; }
    if (t < (size_t)N)     { den2[t] = 0.f; m2[t] = -INFINITY; }
}

// ---- layer 0 GEMMs: fs1 = h @ W1_src, fd1 = h @ W1_dst  (128 -> 128) ------
// block = 256 (8 waves); blockIdx = M-tile; wave w handles N-columns [16w,16w+16)
__global__ __launch_bounds__(256) void gemm_l0(const float* __restrict__ h,
                                               const float* __restrict__ Ws,
                                               const float* __restrict__ Wd,
                                               float* __restrict__ fs,
                                               float* __restrict__ fd) {
    const int wave = threadIdx.x >> 5, lane = threadIdx.x & 31;
    const int m0 = blockIdx.x * 16, n0 = wave * 16;
    const int half = lane >> 4, mrow = lane & 15;
    v8f cs = {}, cd = {};
#pragma unroll
    for (int k0 = 0; k0 < 128; k0 += 32) {
        v16bf a  = load_A(h, m0 + mrow, 128, k0, half);
        v16bf bs = load_B(Ws, 128, k0, n0, lane);
        v16bf bd = load_B(Wd, 128, k0, n0, lane);
        cs = wmma_bf16(a, bs, cs);
        cd = wmma_bf16(a, bd, cd);
    }
    store_D(fs, 128, m0, n0, lane, cs);
    store_D(fd, 128, m0, n0, lane, cd);
}

// ---- layer 1 GEMMs: fs2/fd2/res = h1 @ {W2_src, W2_dst, W_res} (128 -> 16)
__global__ __launch_bounds__(256) void gemm_l1(const float* __restrict__ h1,
                                               const float* __restrict__ Ws,
                                               const float* __restrict__ Wd,
                                               const float* __restrict__ Wr,
                                               float* __restrict__ fs,
                                               float* __restrict__ fd,
                                               float* __restrict__ res,
                                               int mtiles) {
    const int wave = threadIdx.x >> 5, lane = threadIdx.x & 31;
    const int mtile = blockIdx.x * 8 + wave;         // wave-uniform guard
    if (mtile >= mtiles) return;
    const int m0 = mtile * 16;
    const int half = lane >> 4, mrow = lane & 15;
    v8f cs = {}, cd = {}, cr = {};
#pragma unroll
    for (int k0 = 0; k0 < 128; k0 += 32) {
        v16bf a = load_A(h1, m0 + mrow, 128, k0, half);
        cs = wmma_bf16(a, load_B(Ws, 16, k0, 0, lane), cs);
        cd = wmma_bf16(a, load_B(Wd, 16, k0, 0, lane), cd);
        cr = wmma_bf16(a, load_B(Wr, 16, k0, 0, lane), cr);
    }
    store_D(fs, 16, m0, 0, lane, cs);
    store_D(fd, 16, m0, 0, lane, cd);
    store_D(res, 16, m0, 0, lane, cr);
}

// ---- layer 0 edge scores: score[e,h] = a1[h] . lrelu(fs[src]+fd[dst]) -----
// one wave per edge; lanes 0-15 = head 0 dims, lanes 16-31 = head 1 dims
__global__ __launch_bounds__(256) void edge_score_l0(const int* __restrict__ src,
                                                     const int* __restrict__ dst,
                                                     const float* __restrict__ fs,
                                                     const float* __restrict__ fd,
                                                     const float* __restrict__ a1,
                                                     float* __restrict__ score,
                                                     float* __restrict__ m1, long E) {
    const int wave = threadIdx.x >> 5, lane = threadIdx.x & 31;
    const long e = (long)blockIdx.x * 8 + wave;
    if (e >= E) return;
    const int s = src[e], d = dst[e];
    const float4 vs = *(const float4*)(fs + (size_t)s * 128 + lane * 4);
    const float4 vd = *(const float4*)(fd + (size_t)d * 128 + lane * 4);
    const float4 va = *(const float4*)(a1 + lane * 4);
    float p = va.x * lrelu(vs.x + vd.x) + va.y * lrelu(vs.y + vd.y) +
              va.z * lrelu(vs.z + vd.z) + va.w * lrelu(vs.w + vd.w);
    // reduce within each 16-lane half (one head per half)
    p += __shfl_xor(p, 1);
    p += __shfl_xor(p, 2);
    p += __shfl_xor(p, 4);
    p += __shfl_xor(p, 8);
    if ((lane & 15) == 0) {
        const int hh = lane >> 4;
        score[e * 2 + hh] = p;
        atomicMaxF(&m1[(size_t)d * 2 + hh], p);
    }
}

// ---- layer 0 softmax: ex = exp(score - m[dst]); denom[dst] += ex ----------
__global__ __launch_bounds__(256) void edge_exp_l0(const int* __restrict__ dst,
                                                   const float* __restrict__ m1,
                                                   float* __restrict__ score,
                                                   float* __restrict__ den, long E2) {
    const long t = (long)blockIdx.x * 256 + threadIdx.x;
    if (t >= E2) return;
    const long e = t >> 1;
    const int hh = (int)(t & 1);
    const int d = dst[e];
    const float ex = __expf(score[t] - m1[(size_t)d * 2 + hh]);
    score[t] = ex;
    atomicAdd(&den[(size_t)d * 2 + hh], ex);
}

// ---- layer 0 aggregation: out1[dst] += (ex/denom[dst]) * fs[src] ----------
__global__ __launch_bounds__(256) void edge_aggr_l0(const int* __restrict__ src,
                                                    const int* __restrict__ dst,
                                                    const float* __restrict__ fs,
                                                    const float* __restrict__ score,
                                                    const float* __restrict__ den,
                                                    float* __restrict__ out1, long E) {
    const int wave = threadIdx.x >> 5, lane = threadIdx.x & 31;
    const long e = (long)blockIdx.x * 8 + wave;
    if (e >= E) return;
    const int s = src[e], d = dst[e];
    const int hh = lane >> 4;
    const float alpha = score[e * 2 + hh] / den[(size_t)d * 2 + hh];
    const float4 vs = *(const float4*)(fs + (size_t)s * 128 + lane * 4);
    float* o = out1 + (size_t)d * 128 + lane * 4;
    atomicAdd(o + 0, alpha * vs.x);
    atomicAdd(o + 1, alpha * vs.y);
    atomicAdd(o + 2, alpha * vs.z);
    atomicAdd(o + 3, alpha * vs.w);
}

// ---- elu in place: h1 = elu(out1) -----------------------------------------
__global__ __launch_bounds__(256) void elu_k(float* __restrict__ x, size_t n) {
    const size_t t = (size_t)blockIdx.x * 256 + threadIdx.x;
    if (t >= n) return;
    const float v = x[t];
    x[t] = v > 0.f ? v : __expf(v) - 1.f;
}

// ---- layer 1 edge score (1 head, 16 dims): thread per edge ----------------
__global__ __launch_bounds__(256) void edge_score_l1(const int* __restrict__ src,
                                                     const int* __restrict__ dst,
                                                     const float* __restrict__ fs,
                                                     const float* __restrict__ fd,
                                                     const float* __restrict__ a2,
                                                     float* __restrict__ score,
                                                     float* __restrict__ m2, long E) {
    const long e = (long)blockIdx.x * 256 + threadIdx.x;
    if (e >= E) return;
    const int s = src[e], d = dst[e];
    float acc = 0.f;
#pragma unroll
    for (int q = 0; q < 4; ++q) {
        const float4 vs = *(const float4*)(fs + (size_t)s * 16 + q * 4);
        const float4 vd = *(const float4*)(fd + (size_t)d * 16 + q * 4);
        const float4 va = *(const float4*)(a2 + q * 4);
        acc += va.x * lrelu(vs.x + vd.x) + va.y * lrelu(vs.y + vd.y) +
               va.z * lrelu(vs.z + vd.z) + va.w * lrelu(vs.w + vd.w);
    }
    score[e] = acc;
    atomicMaxF(&m2[d], acc);
}

__global__ __launch_bounds__(256) void edge_exp_l1(const int* __restrict__ dst,
                                                   const float* __restrict__ m2,
                                                   float* __restrict__ score,
                                                   float* __restrict__ den, long E) {
    const long e = (long)blockIdx.x * 256 + threadIdx.x;
    if (e >= E) return;
    const int d = dst[e];
    const float ex = __expf(score[e] - m2[d]);
    score[e] = ex;
    atomicAdd(&den[d], ex);
}

// thread per (edge, quarter-row): 4 dims each
__global__ __launch_bounds__(256) void edge_aggr_l1(const int* __restrict__ src,
                                                    const int* __restrict__ dst,
                                                    const float* __restrict__ fs,
                                                    const float* __restrict__ score,
                                                    const float* __restrict__ den,
                                                    float* __restrict__ out2, long E4) {
    const long t = (long)blockIdx.x * 256 + threadIdx.x;
    if (t >= E4) return;
    const long e = t >> 2;
    const int q = (int)(t & 3);
    const int s = src[e], d = dst[e];
    const float alpha = score[e] / den[d];
    const float4 vs = *(const float4*)(fs + (size_t)s * 16 + q * 4);
    float* o = out2 + (size_t)d * 16 + q * 4;
    atomicAdd(o + 0, alpha * vs.x);
    atomicAdd(o + 1, alpha * vs.y);
    atomicAdd(o + 2, alpha * vs.z);
    atomicAdd(o + 3, alpha * vs.w);
}

// ---- final: logits = out2 + res (mean over 1 head = identity) -------------
__global__ __launch_bounds__(256) void final_add(const float* __restrict__ out2,
                                                 const float* __restrict__ res,
                                                 float* __restrict__ logits, size_t n) {
    const size_t t = (size_t)blockIdx.x * 256 + threadIdx.x;
    if (t >= n) return;
    logits[t] = out2[t] + res[t];
}

static inline unsigned cdiv(size_t a, size_t b) { return (unsigned)((a + b - 1) / b); }

extern "C" void kernel_launch(void* const* d_in, const int* in_sizes, int n_in,
                              void* d_out, int out_size, void* d_ws, size_t ws_size,
                              hipStream_t stream) {
    const float* h    = (const float*)d_in[0];
    const int*   src  = (const int*)d_in[1];
    const int*   dst  = (const int*)d_in[2];
    const float* W1s  = (const float*)d_in[3];
    const float* W1d  = (const float*)d_in[4];
    const float* a1   = (const float*)d_in[5];
    const float* W2s  = (const float*)d_in[6];
    const float* W2d  = (const float*)d_in[7];
    const float* a2   = (const float*)d_in[8];
    const float* Wres = (const float*)d_in[9];
    float* logits = (float*)d_out;

    const int  N = in_sizes[0] / 128;   // 50000
    const long E = in_sizes[1];         // 1600000

    // workspace partition (floats)
    float* ws = (float*)d_ws;
    size_t off = 0;
    float* fs1  = ws + off; off += (size_t)N * 128;
    float* fd1  = ws + off; off += (size_t)N * 128;
    float* out1 = ws + off; off += (size_t)N * 128;   // becomes h1 after elu
    float* sc1  = ws + off; off += (size_t)E * 2;
    float* m1   = ws + off; off += (size_t)N * 2;
    float* dn1  = ws + off; off += (size_t)N * 2;
    float* fs2  = ws + off; off += (size_t)N * 16;
    float* fd2  = ws + off; off += (size_t)N * 16;
    float* res  = ws + off; off += (size_t)N * 16;
    float* sc2  = ws + off; off += (size_t)E;
    float* m2   = ws + off; off += (size_t)N;
    float* dn2  = ws + off; off += (size_t)N;
    float* out2 = ws + off; off += (size_t)N * 16;

    const int mtiles = N / 16;

    init_ws<<<cdiv((size_t)N * 128, 256), 256, 0, stream>>>(out1, dn1, m1, out2, dn2, m2, N);

    // layer 0
    gemm_l0<<<mtiles, 256, 0, stream>>>(h, W1s, W1d, fs1, fd1);
    edge_score_l0<<<cdiv((size_t)E, 8), 256, 0, stream>>>(src, dst, fs1, fd1, a1, sc1, m1, E);
    edge_exp_l0<<<cdiv((size_t)E * 2, 256), 256, 0, stream>>>(dst, m1, sc1, dn1, E * 2);
    edge_aggr_l0<<<cdiv((size_t)E, 8), 256, 0, stream>>>(src, dst, fs1, sc1, dn1, out1, E);
    elu_k<<<cdiv((size_t)N * 128, 256), 256, 0, stream>>>(out1, (size_t)N * 128);

    // layer 1
    gemm_l1<<<cdiv(mtiles, 8), 256, 0, stream>>>(out1, W2s, W2d, Wres, fs2, fd2, res, mtiles);
    edge_score_l1<<<cdiv((size_t)E, 256), 256, 0, stream>>>(src, dst, fs2, fd2, a2, sc2, m2, E);
    edge_exp_l1<<<cdiv((size_t)E, 256), 256, 0, stream>>>(dst, m2, sc2, dn2, E);
    edge_aggr_l1<<<cdiv((size_t)E * 4, 256), 256, 0, stream>>>(src, dst, fs2, sc2, dn2, out2, E * 4);

    final_add<<<cdiv((size_t)N * 16, 256), 256, 0, stream>>>(out2, res, logits, (size_t)N * 16);
}
